// QuantizedConv2dWithBN_56453050139307
// MI455X (gfx1250) — matrix operand order repaired and checked by reading, not compiled, not run on hardware
//
#include <hip/hip_runtime.h>
#include <stdint.h>

typedef _Float16 v8h  __attribute__((ext_vector_type(8)));
typedef _Float16 v16h __attribute__((ext_vector_type(16)));
typedef float    v8f  __attribute__((ext_vector_type(8)));

#define NB   32
#define HH   56
#define WW   56
#define CI   128
#define CO   256
#define MTOT (NB*HH*WW)          // 100352
#define KTOT (9*CI)              // 1152
#define BM 128
#define BN 128
#define BK 32
#define LDA 40                   // As row stride in halves (32 + 8 pad)
#define LDB 40                   // Bs row stride in halves
#define KSTEPS (KTOT/BK)         // 36
#define MTILES (MTOT/BM)         // 784

// workspace layout (32-bit words)
#define WS_XMIN 0
#define WS_XMAX 1
#define WS_WMIN 2
#define WS_WMAX 3
#define WS_SCALE 4
#define WS_BIAS  (WS_SCALE + CO)             // 260
#define WS_SUM   (WS_BIAS  + CO)             // 516
#define WS_SUMQ  (WS_SUM + MTILES*2*128)     // 516 + 200704

__device__ __forceinline__ unsigned fordu(float f) {
  unsigned u = __float_as_uint(f);
  return (u & 0x80000000u) ? ~u : (u | 0x80000000u);
}
__device__ __forceinline__ float funord(unsigned u) {
  return __uint_as_float((u & 0x80000000u) ? (u ^ 0x80000000u) : ~u);
}
__device__ __forceinline__ unsigned umn(unsigned a, unsigned b) { return a < b ? a : b; }
__device__ __forceinline__ unsigned umx(unsigned a, unsigned b) { return a > b ? a : b; }

__global__ void k_init(unsigned* wsU) {
  wsU[WS_XMIN] = 0xFFFFFFFFu; wsU[WS_XMAX] = 0u;
  wsU[WS_WMIN] = 0xFFFFFFFFu; wsU[WS_WMAX] = 0u;
}

// global min/max of x (12.8M floats), float4 loads, block reduce, ordered-uint atomics
__global__ void k_minmax_x(const float4* __restrict__ x4, unsigned* wsU, int n4) {
  __shared__ unsigned smn[256], smx[256];
  unsigned lmn = 0xFFFFFFFFu, lmx = 0u;
  for (int i = blockIdx.x * blockDim.x + threadIdx.x; i < n4; i += gridDim.x * blockDim.x) {
    float4 v = x4[i];
    unsigned a = fordu(v.x), b = fordu(v.y), c = fordu(v.z), d = fordu(v.w);
    lmn = umn(lmn, umn(umn(a, b), umn(c, d)));
    lmx = umx(lmx, umx(umx(a, b), umx(c, d)));
  }
  smn[threadIdx.x] = lmn; smx[threadIdx.x] = lmx;
  __syncthreads();
  for (int s = 128; s > 0; s >>= 1) {
    if ((int)threadIdx.x < s) {
      smn[threadIdx.x] = umn(smn[threadIdx.x], smn[threadIdx.x + s]);
      smx[threadIdx.x] = umx(smx[threadIdx.x], smx[threadIdx.x + s]);
    }
    __syncthreads();
  }
  if (threadIdx.x == 0) { atomicMin(&wsU[WS_XMIN], smn[0]); atomicMax(&wsU[WS_XMAX], smx[0]); }
}

// min/max of w_fold = W * scale[f]
__global__ void k_minmax_w(const float4* __restrict__ W4, const float* __restrict__ ws,
                           unsigned* wsU) {
  __shared__ unsigned smn[256], smx[256];
  const int n4 = (9 * CI * CO) / 4;
  unsigned lmn = 0xFFFFFFFFu, lmx = 0u;
  for (int i = blockIdx.x * blockDim.x + threadIdx.x; i < n4; i += gridDim.x * blockDim.x) {
    float4 v = W4[i];
    int f = (i * 4) & (CO - 1);  // f multiple of 4 -> no wrap within group
    float a = v.x * ws[WS_SCALE + f + 0];
    float b = v.y * ws[WS_SCALE + f + 1];
    float c = v.z * ws[WS_SCALE + f + 2];
    float d = v.w * ws[WS_SCALE + f + 3];
    unsigned ea = fordu(a), eb = fordu(b), ec = fordu(c), ed = fordu(d);
    lmn = umn(lmn, umn(umn(ea, eb), umn(ec, ed)));
    lmx = umx(lmx, umx(umx(ea, eb), umx(ec, ed)));
  }
  smn[threadIdx.x] = lmn; smx[threadIdx.x] = lmx;
  __syncthreads();
  for (int s = 128; s > 0; s >>= 1) {
    if ((int)threadIdx.x < s) {
      smn[threadIdx.x] = umn(smn[threadIdx.x], smn[threadIdx.x + s]);
      smx[threadIdx.x] = umx(smx[threadIdx.x], smx[threadIdx.x + s]);
    }
    __syncthreads();
  }
  if (threadIdx.x == 0) { atomicMin(&wsU[WS_WMIN], smn[0]); atomicMax(&wsU[WS_WMAX], smx[0]); }
}

// conv1 implicit GEMM (full precision -> f16 WMMA), fused per-channel sum / sumsq partials
__global__ __launch_bounds__(256) void k_conv_stats(const float* __restrict__ x,
                                                    const float* __restrict__ Wt,
                                                    float* __restrict__ ws) {
  __shared__ _Float16 As[BM * LDA];
  __shared__ _Float16 Bs[BN * LDB];
  __shared__ float redS[16 * 128];
  __shared__ float redQ[16 * 128];

  const int tid = threadIdx.x;
  const int mtile = blockIdx.x, ftile = blockIdx.y;
  const int lane = tid & 31, wid = tid >> 5;
  const int wm = wid & 3, wn = wid >> 2;
  const int r = lane & 15, hf = lane >> 4;

  // A tile loading coords: 2 threads per pixel, 16 channels each
  const int arow = tid >> 1;
  const int ac = (tid & 1) * 16;
  const int m = mtile * BM + arow;
  const int img = m / (HH * WW);
  const int rem = m % (HH * WW);
  const int oy = rem / WW, ox = rem % WW;

  // B tile loading coords: thread -> (k row, 16 f columns)
  const int bk = tid >> 3;
  const int bf = (tid & 7) * 16;
  const int fbase = ftile * BN;

  v8f acc[2][4];
  #pragma unroll
  for (int i = 0; i < 2; ++i)
    #pragma unroll
    for (int j = 0; j < 4; ++j)
      #pragma unroll
      for (int v = 0; v < 8; ++v) acc[i][j][v] = 0.0f;

  for (int ks = 0; ks < KSTEPS; ++ks) {
    const int tap = ks >> 2;
    const int c0 = (ks & 3) << 5;
    const int ky = tap / 3, kx = tap % 3;
    const int iy = oy + ky - 1, ix = ox + kx - 1;

    __syncthreads();
    {  // A tile: 128 pixels x 32 K (pad -> zeros)
      float av[16];
      if ((unsigned)iy < (unsigned)HH && (unsigned)ix < (unsigned)WW) {
        const float4* p =
            (const float4*)(x + (size_t)(((img * HH + iy) * WW + ix)) * CI + c0 + ac);
        float4 t0 = p[0], t1 = p[1], t2 = p[2], t3 = p[3];
        av[0] = t0.x; av[1] = t0.y; av[2] = t0.z; av[3] = t0.w;
        av[4] = t1.x; av[5] = t1.y; av[6] = t1.z; av[7] = t1.w;
        av[8] = t2.x; av[9] = t2.y; av[10] = t2.z; av[11] = t2.w;
        av[12] = t3.x; av[13] = t3.y; av[14] = t3.z; av[15] = t3.w;
      } else {
        #pragma unroll
        for (int j = 0; j < 16; ++j) av[j] = 0.0f;
      }
      v8h h0, h1;
      #pragma unroll
      for (int j = 0; j < 8; ++j) { h0[j] = (_Float16)av[j]; h1[j] = (_Float16)av[j + 8]; }
      *(v8h*)(&As[arow * LDA + ac]) = h0;
      *(v8h*)(&As[arow * LDA + ac + 8]) = h1;
    }
    {  // B tile: 32 K x 128 F, stored transposed Bs[f][k]
      const float4* p =
          (const float4*)(Wt + (size_t)((tap * CI + c0 + bk)) * CO + fbase + bf);
      float4 t0 = p[0], t1 = p[1], t2 = p[2], t3 = p[3];
      float bv[16] = {t0.x, t0.y, t0.z, t0.w, t1.x, t1.y, t1.z, t1.w,
                      t2.x, t2.y, t2.z, t2.w, t3.x, t3.y, t3.z, t3.w};
      #pragma unroll
      for (int j = 0; j < 16; ++j) Bs[(bf + j) * LDB + bk] = (_Float16)bv[j];
    }
    __syncthreads();

    v16h a[2], b[4];
    #pragma unroll
    for (int i = 0; i < 2; ++i) {
      const _Float16* ap = &As[(wm * 32 + i * 16 + r) * LDA];
      v8h lo = *(const v8h*)(ap + hf * 8);
      v8h hi = *(const v8h*)(ap + 16 + hf * 8);
      a[i] = __builtin_shufflevector(lo, hi, 0, 1, 2, 3, 4, 5, 6, 7, 8, 9, 10, 11, 12, 13, 14, 15);
    }
    #pragma unroll
    for (int j = 0; j < 4; ++j) {
      const _Float16* bp = &Bs[(wn * 64 + j * 16 + r) * LDB];
      v8h lo = *(const v8h*)(bp + hf * 8);
      v8h hi = *(const v8h*)(bp + 16 + hf * 8);
      b[j] = __builtin_shufflevector(lo, hi, 0, 1, 2, 3, 4, 5, 6, 7, 8, 9, 10, 11, 12, 13, 14, 15);
    }
    #pragma unroll
    for (int i = 0; i < 2; ++i)
      #pragma unroll
      for (int j = 0; j < 4; ++j)
        acc[i][j] = __builtin_amdgcn_wmma_f32_16x16x32_f16(
            false, a[i], false, b[j], (short)0, acc[i][j], false, false);
  }

  // deterministic per-block per-channel sum/sumsq: slot = (wm,i,hf) -> single writer per (slot,col)
  #pragma unroll
  for (int i = 0; i < 2; ++i)
    #pragma unroll
    for (int j = 0; j < 4; ++j) {
      float s = 0.0f, q = 0.0f;
      #pragma unroll
      for (int v = 0; v < 8; ++v) { float t = acc[i][j][v]; s += t; q += t * t; }
      int col = wn * 64 + j * 16 + r;
      int slot = wm * 4 + i * 2 + hf;
      redS[slot * 128 + col] = s;
      redQ[slot * 128 + col] = q;
    }
  __syncthreads();
  if (tid < 128) {
    float s = 0.0f, q = 0.0f;
    #pragma unroll
    for (int k = 0; k < 16; ++k) { s += redS[k * 128 + tid]; q += redQ[k * 128 + tid]; }
    ws[WS_SUM + (mtile * 2 + ftile) * 128 + tid] = s;
    ws[WS_SUMQ + (mtile * 2 + ftile) * 128 + tid] = q;
  }
}

// reduce partials -> mean/var -> scale = gamma*inv_std, bias_fold = beta - gamma*mean*inv_std
__global__ void k_stats(float* ws, const float* __restrict__ gamma,
                        const float* __restrict__ beta) {
  const int f = threadIdx.x;  // 256 threads
  const int ftile = f >> 7, loc = f & 127;
  double s = 0.0, q = 0.0;
  for (int mt = 0; mt < MTILES; ++mt) {
    int idx = (mt * 2 + ftile) * 128 + loc;
    s += (double)ws[WS_SUM + idx];
    q += (double)ws[WS_SUMQ + idx];
  }
  double mean = s / (double)MTOT;
  double var = q / (double)MTOT - mean * mean;
  double inv = 1.0 / sqrt(var + 1e-5);
  float g = gamma[f];
  ws[WS_SCALE + f] = (float)((double)g * inv);
  ws[WS_BIAS + f] = beta[f] - (float)((double)g * mean * inv);
}

// conv2: quantize x and w_fold on the fly, WMMA, + bias_fold, ReLU
__global__ __launch_bounds__(256) void k_conv_out(const float* __restrict__ x,
                                                  const float* __restrict__ Wt,
                                                  const float* __restrict__ ws,
                                                  float* __restrict__ out) {
  __shared__ _Float16 As[BM * LDA];
  __shared__ _Float16 Bs[BN * LDB];

  const int tid = threadIdx.x;
  const int mtile = blockIdx.x, ftile = blockIdx.y;
  const int lane = tid & 31, wid = tid >> 5;
  const int wm = wid & 3, wn = wid >> 2;
  const int r = lane & 15, hf = lane >> 4;

  const int arow = tid >> 1;
  const int ac = (tid & 1) * 16;
  const int m = mtile * BM + arow;
  const int img = m / (HH * WW);
  const int rem = m % (HH * WW);
  const int oy = rem / WW, ox = rem % WW;

  const int bk = tid >> 3;
  const int bf = (tid & 7) * 16;
  const int fbase = ftile * BN;

  // quantization scalars (written by earlier kernels on this stream)
  const unsigned* wsU = (const unsigned*)ws;
  const float amn = funord(wsU[WS_XMIN]);
  const float amx = funord(wsU[WS_XMAX]);
  const float astep = (amx - amn) * (1.0f / 255.0f);
  const float arcp = 1.0f / astep;
  const float wmn = funord(wsU[WS_WMIN]);
  const float wmx = funord(wsU[WS_WMAX]);
  const float wstep = (wmx - wmn) * (1.0f / 255.0f);
  const float wrcp = 1.0f / wstep;

  float scv[16];
  #pragma unroll
  for (int j = 0; j < 16; ++j) scv[j] = ws[WS_SCALE + fbase + bf + j];

  v8f acc[2][4];
  #pragma unroll
  for (int i = 0; i < 2; ++i)
    #pragma unroll
    for (int j = 0; j < 4; ++j)
      #pragma unroll
      for (int v = 0; v < 8; ++v) acc[i][j][v] = 0.0f;

  for (int ks = 0; ks < KSTEPS; ++ks) {
    const int tap = ks >> 2;
    const int c0 = (ks & 3) << 5;
    const int ky = tap / 3, kx = tap % 3;
    const int iy = oy + ky - 1, ix = ox + kx - 1;

    __syncthreads();
    {  // A tile: fake-quantized activations (padding stays exact 0)
      float av[16];
      if ((unsigned)iy < (unsigned)HH && (unsigned)ix < (unsigned)WW) {
        const float4* p =
            (const float4*)(x + (size_t)(((img * HH + iy) * WW + ix)) * CI + c0 + ac);
        float4 t0 = p[0], t1 = p[1], t2 = p[2], t3 = p[3];
        av[0] = t0.x; av[1] = t0.y; av[2] = t0.z; av[3] = t0.w;
        av[4] = t1.x; av[5] = t1.y; av[6] = t1.z; av[7] = t1.w;
        av[8] = t2.x; av[9] = t2.y; av[10] = t2.z; av[11] = t2.w;
        av[12] = t3.x; av[13] = t3.y; av[14] = t3.z; av[15] = t3.w;
        #pragma unroll
        for (int j = 0; j < 16; ++j)
          av[j] = rintf((av[j] - amn) * arcp) * astep + amn;
      } else {
        #pragma unroll
        for (int j = 0; j < 16; ++j) av[j] = 0.0f;
      }
      v8h h0, h1;
      #pragma unroll
      for (int j = 0; j < 8; ++j) { h0[j] = (_Float16)av[j]; h1[j] = (_Float16)av[j + 8]; }
      *(v8h*)(&As[arow * LDA + ac]) = h0;
      *(v8h*)(&As[arow * LDA + ac + 8]) = h1;
    }
    {  // B tile: fold + fake-quantize weights, stored transposed
      const float4* p =
          (const float4*)(Wt + (size_t)((tap * CI + c0 + bk)) * CO + fbase + bf);
      float4 t0 = p[0], t1 = p[1], t2 = p[2], t3 = p[3];
      float bv[16] = {t0.x, t0.y, t0.z, t0.w, t1.x, t1.y, t1.z, t1.w,
                      t2.x, t2.y, t2.z, t2.w, t3.x, t3.y, t3.z, t3.w};
      #pragma unroll
      for (int j = 0; j < 16; ++j) {
        float wf = bv[j] * scv[j];
        wf = rintf((wf - wmn) * wrcp) * wstep + wmn;
        Bs[(bf + j) * LDB + bk] = (_Float16)wf;
      }
    }
    __syncthreads();

    v16h a[2], b[4];
    #pragma unroll
    for (int i = 0; i < 2; ++i) {
      const _Float16* ap = &As[(wm * 32 + i * 16 + r) * LDA];
      v8h lo = *(const v8h*)(ap + hf * 8);
      v8h hi = *(const v8h*)(ap + 16 + hf * 8);
      a[i] = __builtin_shufflevector(lo, hi, 0, 1, 2, 3, 4, 5, 6, 7, 8, 9, 10, 11, 12, 13, 14, 15);
    }
    #pragma unroll
    for (int j = 0; j < 4; ++j) {
      const _Float16* bp = &Bs[(wn * 64 + j * 16 + r) * LDB];
      v8h lo = *(const v8h*)(bp + hf * 8);
      v8h hi = *(const v8h*)(bp + 16 + hf * 8);
      b[j] = __builtin_shufflevector(lo, hi, 0, 1, 2, 3, 4, 5, 6, 7, 8, 9, 10, 11, 12, 13, 14, 15);
    }
    #pragma unroll
    for (int i = 0; i < 2; ++i)
      #pragma unroll
      for (int j = 0; j < 4; ++j)
        acc[i][j] = __builtin_amdgcn_wmma_f32_16x16x32_f16(
            false, a[i], false, b[j], (short)0, acc[i][j], false, false);
  }

  float biasj[4];
  #pragma unroll
  for (int j = 0; j < 4; ++j) biasj[j] = ws[WS_BIAS + fbase + wn * 64 + j * 16 + r];

  #pragma unroll
  for (int i = 0; i < 2; ++i)
    #pragma unroll
    for (int j = 0; j < 4; ++j) {
      const int fcol = fbase + wn * 64 + j * 16 + r;
      #pragma unroll
      for (int v = 0; v < 8; ++v) {
        const int mrow = mtile * BM + wm * 32 + i * 16 + hf * 8 + v;
        out[(size_t)mrow * CO + fcol] = fmaxf(acc[i][j][v] + biasj[j], 0.0f);
      }
    }
}

extern "C" void kernel_launch(void* const* d_in, const int* in_sizes, int n_in,
                              void* d_out, int out_size, void* d_ws, size_t ws_size,
                              hipStream_t stream) {
  (void)in_sizes; (void)n_in; (void)out_size; (void)ws_size;
  const float* x = (const float*)d_in[0];
  const float* Wt = (const float*)d_in[1];
  const float* gamma = (const float*)d_in[2];
  const float* beta = (const float*)d_in[3];
  float* out = (float*)d_out;
  float* ws = (float*)d_ws;
  unsigned* wsU = (unsigned*)d_ws;

  k_init<<<1, 1, 0, stream>>>(wsU);
  k_minmax_x<<<1024, 256, 0, stream>>>((const float4*)x, wsU, (NB * HH * WW * CI) / 4);
  k_conv_stats<<<dim3(MTILES, 2), 256, 0, stream>>>(x, Wt, ws);
  k_stats<<<1, 256, 0, stream>>>(ws, gamma, beta);
  k_minmax_w<<<288, 256, 0, stream>>>((const float4*)Wt, ws, wsU);
  k_conv_out<<<dim3(MTILES, 2), 256, 0, stream>>>(x, Wt, ws, out);
}
